// LSTM_58763742544664
// MI455X (gfx1250) — matrix operand order repaired and checked by reading, not compiled, not run on hardware
//
#include <hip/hip_runtime.h>

// LSTM on MI455X (gfx1250, wave32, WMMA).
// - batch-parallel recurrence: 8 blocks x 16 batch rows, zero cross-WG sync
// - f16 WMMA 16x16x32, f32 accumulation; c-state in VGPRs for all 1024 steps
// - h_{t} ping-pongs through LDS (ds_load_b128 A-frags); x_t tile double-
//   buffered in LDS via GLOBAL_LOAD_ASYNC_TO_LDS_B128, prefetched 1 step ahead
// - per k-chunk: 8 B-frags batch-loaded (clause of 16 b128), then 8 WMMAs

#define Bv 128
#define Tv 1024
#define Iv 64
#define Hv 512
#define Gv 2048   // 4*H
#define MB 16     // batch rows per block

typedef __attribute__((ext_vector_type(16))) _Float16 v16h;
typedef __attribute__((ext_vector_type(8)))  _Float16 v8h;
typedef __attribute__((ext_vector_type(8)))  float    v8f;

// A 16x32 f16 (ISA 7.12.2): lane L -> row M=L&15; halves at K kb..kb+7 and
// kb+16..kb+23, kb=(L>>4)*8.  base must be derived DIRECTLY from the __shared__
// array (single pointer + int offsets) so clang infers addrspace(3) -> ds_load.
__device__ __forceinline__ v16h fragA_off(const _Float16* base, int off) {
  union { v16h v; v8h h[2]; } u;
  u.h[0] = *(const v8h*)(base + off);
  u.h[1] = *(const v8h*)(base + off + 16);
  return u.v;
}
// B 32x16 f16: lane L -> col N=L&15; K kb..kb+7 and kb+16..kb+23, kb=(L>>4)*8.
// W row-major (N x K) == B^T; rowptr = W + N*ldb + (L>>4)*8.
__device__ __forceinline__ v16h fragB_from(const _Float16* rowptr, int kc) {
  union { v16h v; v8h h[2]; } u;
  u.h[0] = *(const v8h*)(rowptr + kc);
  u.h[1] = *(const v8h*)(rowptr + kc + 16);
  return u.v;
}

__device__ __forceinline__ float sigmoidf_(float x) {
  return 1.0f / (1.0f + __expf(-x));
}
__device__ __forceinline__ float tanhf_(float x) {
  float t = __expf(-2.0f * fabsf(x));      // in (0,1], no overflow
  float r = (1.0f - t) / (1.0f + t);
  return copysignf(r, x);
}

// async stage: 16B per lane-op, global -> LDS, tracked by ASYNCcnt
__device__ __forceinline__ void stage_async(const _Float16* gsrc, long ldx,
                                            _Float16* ldst, int Kin, int rsh,
                                            int tid) {
  const int kin8  = Kin >> 3;              // 16B chunks per row (8 or 64)
  const int total = MB * kin8;             // 128 or 1024 chunks
  for (int i = tid; i < total; i += 512) {
    int row = i >> rsh;
    int c8  = i & (kin8 - 1);
    unsigned long long ga =
        (unsigned long long)(uintptr_t)(gsrc + (long)row * ldx + (c8 << 3));
    unsigned lo = (unsigned)(uintptr_t)(ldst + row * Kin + (c8 << 3));
    asm volatile("global_load_async_to_lds_b128 %0, %1, off"
                 :: "v"(lo), "v"(ga) : "memory");
  }
}

// ---- elementwise helpers ----------------------------------------------------
__global__ void k_cast_f16(const float* __restrict__ in, _Float16* __restrict__ out, long n) {
  long i = (long)blockIdx.x * blockDim.x + threadIdx.x;
  long stride = (long)gridDim.x * blockDim.x;
  for (; i < n; i += stride) out[i] = (_Float16)in[i];
}

__global__ void k_addbias(const float* __restrict__ a, const float* __restrict__ b,
                          float* __restrict__ o, int n) {
  int i = blockIdx.x * blockDim.x + threadIdx.x;
  if (i < n) o[i] = a[i] + b[i];
}

__global__ void k_h0(const float* __restrict__ cond, const float* __restrict__ Wc,
                     const float* __restrict__ bc,   const float* __restrict__ Wi,
                     const float* __restrict__ bi,   float* __restrict__ h0) {
  int idx = blockIdx.x * blockDim.x + threadIdx.x;
  if (idx >= Bv * Hv) return;
  int b = idx / Hv, h = idx % Hv;
  float acc = bi[h];
  #pragma unroll
  for (int co = 0; co < 32; ++co) {
    float cp = bc[co];
    #pragma unroll
    for (int ci = 0; ci < 10; ++ci) cp += cond[b * 10 + ci] * Wc[co * 10 + ci];
    acc += cp * Wi[h * 32 + co];
  }
  h0[idx] = acc;
}

__global__ void k_out(const float* __restrict__ hlast, const float* __restrict__ Wo,
                      const float* __restrict__ bo, float* __restrict__ y) {
  int idx = blockIdx.x * blockDim.x + threadIdx.x;
  if (idx >= Bv * 16) return;
  int b = idx >> 4, o = idx & 15;
  float acc = bo[o];
  for (int k = 0; k < Hv; ++k) acc += hlast[(long)b * Hv + k] * Wo[o * Hv + k];
  y[idx] = acc;
}

// ---- fused LSTM layer -------------------------------------------------------
// Wave w owns 32 h-columns; computes 8 gate tiles (4 gates x 2 col-tiles) so
// i/f/g/o for a given (b,j) land in the same lane -> c stays in VGPRs.
// Tile index i = g*2 + jt : gate g = i>>1, col-tile jt = i&1.
__global__ __launch_bounds__(512) void k_lstm_fused(
    const _Float16* __restrict__ xin, int Kin, int rsh, // [B][T][Kin]; rsh=log2(Kin/8)
    const _Float16* __restrict__ Wih,                   // [2048][Kin]
    const _Float16* __restrict__ Whh,                   // [2048][512]
    const float*    __restrict__ bias,                  // [2048] = bih + bhh
    const float*    __restrict__ h0,                    // [B][512]
    _Float16*       __restrict__ hout,                  // [B][T][512] (may alias xin)
    float*          __restrict__ hlast)                 // [B][512]
{
  __shared__ __align__(16) _Float16 hbuf[2 * MB * Hv];  // 32 KB h ping-pong
  __shared__ __align__(16) _Float16 xtile[2 * MB * Hv]; // 32 KB x_t double buffer
  const int tid  = threadIdx.x;
  const int lane = tid & 31;
  const int wave = tid >> 5;            // 0..15
  const int b0   = blockIdx.x * MB;
  const int j0   = wave * 32;
  const long ldx = (long)Tv * Kin;
  const _Float16* xbase = xin + (long)b0 * Tv * Kin;

  const _Float16* hflat = &hbuf[0];     // single base pointers (keep LDS
  const _Float16* xflat = &xtile[0];    //  addrspace inference intact)

  for (int i = tid; i < MB * Hv; i += 512)
    hbuf[i] = (_Float16)h0[(long)(b0 + (i >> 9)) * Hv + (i & 511)];

  // prologue: stage x-tile for t=0 into buffer 0
  stage_async(xbase, ldx, &xtile[0], Kin, rsh, tid);
  asm volatile("s_wait_asynccnt 0x0" ::: "memory");
  __syncthreads();

  // t-invariant per-lane values
  const int laneA = lane & 15;
  const int kbA   = (lane >> 4) << 3;
  const int kbB   = (lane >> 4) << 3;
  const int aoffX = laneA * Kin + kbA;  // A-frag offset within one x buffer
  const int aoffH = laneA * Hv  + kbA;  // A-frag offset within one h buffer
  const _Float16* bihp[8];
  const _Float16* bhhp[8];
  float bcol[8];
  #pragma unroll
  for (int i = 0; i < 8; ++i) {
    int n = (i >> 1) * Hv + j0 + (i & 1) * 16 + laneA;
    bihp[i] = Wih + (long)n * Kin + kbB;
    bhhp[i] = Whh + (long)n * Hv  + kbB;
    bcol[i] = bias[n];
  }

  float creg[2][8];
  #pragma unroll
  for (int jt = 0; jt < 2; ++jt)
    #pragma unroll
    for (int v = 0; v < 8; ++v) creg[jt][v] = 0.0f;

  for (int t = 0; t < Tv; ++t) {
    // prefetch x-tile for t+1 (overlaps this whole step's compute)
    if (t + 1 < Tv)
      stage_async(xbase + (long)(t + 1) * Kin, ldx,
                  &xtile[((t + 1) & 1) * (MB * Hv)], Kin, rsh, tid);

    v8f acc8[8];
    #pragma unroll
    for (int i = 0; i < 8; ++i)
      #pragma unroll
      for (int v = 0; v < 8; ++v) acc8[i][v] = bcol[i];

    // fused input projection: gates += x_t @ Wih^T  (A from LDS x-tile)
    const int xoff = ((t & 1) ? MB * Hv : 0) + aoffX;
    for (int kc = 0; kc < Kin; kc += 32) {
      v16h bf[8];
      #pragma unroll
      for (int i = 0; i < 8; ++i) bf[i] = fragB_from(bihp[i], kc);
      v16h a = fragA_off(xflat, xoff + kc);
      #pragma unroll
      for (int i = 0; i < 8; ++i)
        acc8[i] = __builtin_amdgcn_wmma_f32_16x16x32_f16(
            false, a, false, bf[i], (short)0, acc8[i], false, false);
    }

    // recurrent projection: gates += h_{t-1} @ Whh^T  (A from LDS h-buffer)
    const int hoff = ((t & 1) ? MB * Hv : 0) + aoffH;
    #pragma unroll 2
    for (int kc = 0; kc < Hv; kc += 32) {
      v16h bf[8];
      #pragma unroll
      for (int i = 0; i < 8; ++i) bf[i] = fragB_from(bhhp[i], kc);
      v16h a = fragA_off(hflat, hoff + kc);
      #pragma unroll
      for (int i = 0; i < 8; ++i)
        acc8[i] = __builtin_amdgcn_wmma_f32_16x16x32_f16(
            false, a, false, bf[i], (short)0, acc8[i], false, false);
    }

    // gates -> c,h (per-lane: D layout puts i/f/g/o for (m,j) in one lane)
    const int mbase = (lane >> 4) << 3;
    const int wbase = (((t + 1) & 1) ? MB * Hv : 0);
    #pragma unroll
    for (int jt = 0; jt < 2; ++jt) {
      const int j = j0 + jt * 16 + laneA;
      #pragma unroll
      for (int v = 0; v < 8; ++v) {
        float ig = sigmoidf_(acc8[0 + jt][v]);
        float fg = sigmoidf_(acc8[2 + jt][v]);
        float gg = tanhf_(acc8[4 + jt][v]);
        float og = sigmoidf_(acc8[6 + jt][v]);
        float c  = fg * creg[jt][v] + ig * gg;
        creg[jt][v] = c;
        float h  = og * tanhf_(c);
        int   m  = mbase + v;
        _Float16 hh = (_Float16)h;
        hbuf[wbase + m * Hv + j] = hh;
        hout[((long)(b0 + m) * Tv + t) * Hv + j] = hh;
        if (t == Tv - 1) hlast[(long)(b0 + m) * Hv + j] = h;
      }
    }
    // drain this wave's async x-tile stores, then make h_t + x_{t+1} visible
    asm volatile("s_wait_asynccnt 0x0" ::: "memory");
    __syncthreads();
  }
}

// ---- launcher ---------------------------------------------------------------
extern "C" void kernel_launch(void* const* d_in, const int* in_sizes, int n_in,
                              void* d_out, int out_size, void* d_ws, size_t ws_size,
                              hipStream_t stream) {
  const float* x    = (const float*)d_in[0];
  const float* cond = (const float*)d_in[1];
  const float* Wc   = (const float*)d_in[2];
  const float* bc   = (const float*)d_in[3];
  const float* Wi   = (const float*)d_in[4];
  const float* bi   = (const float*)d_in[5];
  const float* Wih0 = (const float*)d_in[6];
  const float* Whh0 = (const float*)d_in[7];
  const float* bih0 = (const float*)d_in[8];
  const float* bhh0 = (const float*)d_in[9];
  const float* Wih1 = (const float*)d_in[10];
  const float* Whh1 = (const float*)d_in[11];
  const float* bih1 = (const float*)d_in[12];
  const float* bhh1 = (const float*)d_in[13];
  const float* Wo   = (const float*)d_in[14];
  const float* bo   = (const float*)d_in[15];
  float* y = (float*)d_out;

  char* ws = (char*)d_ws;
  size_t off = 0;
  auto take = [&](size_t bytes) {
    off = (off + 255) & ~(size_t)255;
    size_t o = off; off += bytes; return o;
  };
  _Float16* xf    = (_Float16*)(ws + take((size_t)Bv * Tv * Iv * 2)); // 16 MB
  _Float16* wih0f = (_Float16*)(ws + take((size_t)Gv * Iv * 2));
  _Float16* whh0f = (_Float16*)(ws + take((size_t)Gv * Hv * 2));
  _Float16* wih1f = (_Float16*)(ws + take((size_t)Gv * Hv * 2));
  _Float16* whh1f = (_Float16*)(ws + take((size_t)Gv * Hv * 2));
  float*    bias0 = (float*)(ws + take((size_t)Gv * 4));
  float*    bias1 = (float*)(ws + take((size_t)Gv * 4));
  float*    h0b   = (float*)(ws + take((size_t)Bv * Hv * 4));
  float*    hlast = (float*)(ws + take((size_t)Bv * Hv * 4));
  _Float16* houtf = (_Float16*)(ws + take((size_t)Bv * Tv * Hv * 2)); // 128 MB

  k_cast_f16<<<2048, 256, 0, stream>>>(x,    xf,    (long)Bv * Tv * Iv);
  k_cast_f16<<<512,  256, 0, stream>>>(Wih0, wih0f, (long)Gv * Iv);
  k_cast_f16<<<2048, 256, 0, stream>>>(Whh0, whh0f, (long)Gv * Hv);
  k_cast_f16<<<2048, 256, 0, stream>>>(Wih1, wih1f, (long)Gv * Hv);
  k_cast_f16<<<2048, 256, 0, stream>>>(Whh1, whh1f, (long)Gv * Hv);

  k_addbias<<<(Gv + 255) / 256, 256, 0, stream>>>(bih0, bhh0, bias0, Gv);
  k_addbias<<<(Gv + 255) / 256, 256, 0, stream>>>(bih1, bhh1, bias1, Gv);

  k_h0<<<(Bv * Hv + 255) / 256, 256, 0, stream>>>(cond, Wc, bc, Wi, bi, h0b);

  // layer 0: x(f16) -> houtf ; layer 1: houtf in-place (async read of row t+1
  // completes before the step-t barrier; row t+1 written only in step t+1)
  k_lstm_fused<<<Bv / MB, 512, 0, stream>>>(xf,    Iv, 3, wih0f, whh0f, bias0, h0b, houtf, hlast);
  k_lstm_fused<<<Bv / MB, 512, 0, stream>>>(houtf, Hv, 6, wih1f, whh1f, bias1, h0b, houtf, hlast);

  k_out<<<(Bv * 16 + 255) / 256, 256, 0, stream>>>(hlast, Wo, bo, y);
}
